// LocalAttention_77635828843044
// MI455X (gfx1250) — matrix operand order repaired and checked by reading, not compiled, not run on hardware
//
#include <hip/hip_runtime.h>

typedef __attribute__((ext_vector_type(16))) __bf16 v16bf;
typedef __attribute__((ext_vector_type(8)))  __bf16 v8bf;
typedef __attribute__((ext_vector_type(8)))  float  v8f;

namespace {
constexpr int NB = 4;      // batch
constexpr int NL = 2048;   // sequence length
constexpr int ND = 512;    // model dim
constexpr int NH = 8;      // heads
constexpr int DH = 64;     // head dim
constexpr float SCALE = 0.125f;              // 1/sqrt(64)
constexpr float L2E   = 1.44269504088896f;   // log2(e)
}

// D = A(16x32 bf16) * B(32x16 bf16) + C(16x16 f32)
__device__ __forceinline__ v8f wmma_bf16(v16bf a, v16bf b, v8f c) {
  return __builtin_amdgcn_wmma_f32_16x16x32_bf16(false, a, false, b, (short)0, c,
                                                 false, false);
}

// Assemble a 32-byte WMMA operand fragment from two 16-byte global loads.
__device__ __forceinline__ v16bf ld_frag(const __bf16* p0, const __bf16* p1) {
  v16bf f;
  ((uint4*)&f)[0] = *(const uint4*)p0;
  ((uint4*)&f)[1] = *(const uint4*)p1;
  return f;
}

// ---------------------------------------------------------------------------
// fp32 -> bf16 elementwise convert (weights)
// ---------------------------------------------------------------------------
__global__ void k_cvt(const float* __restrict__ src, __bf16* __restrict__ dst,
                      int n) {
  for (int i = blockIdx.x * blockDim.x + threadIdx.x; i < n;
       i += gridDim.x * blockDim.x)
    dst[i] = (__bf16)src[i];
}

// ---------------------------------------------------------------------------
// (B, C=512, L) f32  ->  (B, L, C=512) bf16  via LDS tile transpose
// ---------------------------------------------------------------------------
__global__ void __launch_bounds__(256) k_transpose(const float* __restrict__ x,
                                                   __bf16* __restrict__ xt) {
  __shared__ float tile[32][33];
  const int b = blockIdx.z;
  const int c0 = blockIdx.y * 32;
  const int l0 = blockIdx.x * 32;
  const int tx = threadIdx.x, ty = threadIdx.y;
  const float* xp = x + (size_t)b * ND * NL;
#pragma unroll
  for (int i = 0; i < 32; i += 8)
    tile[ty + i][tx] = xp[(size_t)(c0 + ty + i) * NL + l0 + tx];
  __syncthreads();
  __bf16* op = xt + (size_t)b * NL * ND;
#pragma unroll
  for (int i = 0; i < 32; i += 8)
    op[(size_t)(l0 + ty + i) * ND + c0 + tx] = (__bf16)tile[tx][ty + i];
}

// ---------------------------------------------------------------------------
// Q/K projection: Y = W(512x512) @ X + b, output as (B, H, L, DH) bf16
// A = W (M=outch x K=inch), B = Xt (N=pos x K=inch).
// Per wave: 32x64 tile (2 M-subtiles share every B fragment).
// ---------------------------------------------------------------------------
__global__ void __launch_bounds__(128) k_proj_qk(
    const __bf16* __restrict__ Wb, const __bf16* __restrict__ Xt,
    const float* __restrict__ bias, __bf16* __restrict__ out) {
  const int lane = threadIdx.x & 31;
  const int gw = blockIdx.x * 4 + (threadIdx.x >> 5);
  const int b = gw >> 9;          // 512 waves per batch
  const int rem = gw & 511;
  const int m0 = (rem >> 5) * 32;  // 16 channel tiles of 32
  const int n0 = (rem & 31) * 64;  // 32 position tiles of 64
  const bool hi = lane >= 16;
  const int off = hi ? 8 : 0;
  const int klo = hi ? 16 : 0;
  const int ln = lane & 15;

  v8f acc[2][4] = {};
  for (int kk = 0; kk < ND; kk += 32) {
    v16bf af[2];
#pragma unroll
    for (int s = 0; s < 2; ++s) {
      const __bf16* ap = Wb + (size_t)(m0 + s * 16 + ln) * ND + kk;
      af[s] = ld_frag(ap + klo, ap + klo + 8);
    }
#pragma unroll
    for (int t = 0; t < 4; ++t) {
      const __bf16* bp =
          Xt + ((size_t)b * NL + n0 + t * 16 + ln) * ND + kk + klo;
      v16bf bfr = ld_frag(bp, bp + 8);
      acc[0][t] = wmma_bf16(af[0], bfr, acc[0][t]);
      acc[1][t] = wmma_bf16(af[1], bfr, acc[1][t]);
    }
  }
  const int h = m0 >> 6;  // whole 32-row tile sits inside one head block
#pragma unroll
  for (int s = 0; s < 2; ++s) {
    const int mrow = m0 + s * 16 + off;
    const int dh0 = mrow & 63;
    float bv[8];
#pragma unroll
    for (int r = 0; r < 8; ++r) bv[r] = bias[mrow + r];
#pragma unroll
    for (int t = 0; t < 4; ++t) {
      const int pos = n0 + t * 16 + ln;
      v8bf o;
#pragma unroll
      for (int r = 0; r < 8; ++r) o[r] = (__bf16)(acc[s][t][r] + bv[r]);
      *(uint4*)(out + (((size_t)b * NH + h) * NL + pos) * DH + dh0) =
          *(uint4*)&o;
    }
  }
}

// ---------------------------------------------------------------------------
// V projection computed transposed: output as (B, H, DH, L) bf16.
// A = Xt (M=pos x K=inch), B = W (N=outch x K=inch). Per wave: 32x64 tile.
// ---------------------------------------------------------------------------
__global__ void __launch_bounds__(128) k_proj_v(
    const __bf16* __restrict__ Wb, const __bf16* __restrict__ Xt,
    const float* __restrict__ bias, __bf16* __restrict__ out) {
  const int lane = threadIdx.x & 31;
  const int gw = blockIdx.x * 4 + (threadIdx.x >> 5);
  const int b = gw >> 9;
  const int rem = gw & 511;
  const int m0 = (rem >> 3) * 32;  // 64 position tiles of 32
  const int n0 = (rem & 7) * 64;   // 8 channel tiles of 64
  const bool hi = lane >= 16;
  const int off = hi ? 8 : 0;
  const int klo = hi ? 16 : 0;
  const int ln = lane & 15;

  v8f acc[2][4] = {};
  for (int kk = 0; kk < ND; kk += 32) {
    v16bf af[2];
#pragma unroll
    for (int s = 0; s < 2; ++s) {
      const __bf16* ap = Xt + ((size_t)b * NL + m0 + s * 16 + ln) * ND + kk;
      af[s] = ld_frag(ap + klo, ap + klo + 8);
    }
#pragma unroll
    for (int t = 0; t < 4; ++t) {
      const __bf16* bp = Wb + (size_t)(n0 + t * 16 + ln) * ND + kk + klo;
      v16bf bfr = ld_frag(bp, bp + 8);
      acc[0][t] = wmma_bf16(af[0], bfr, acc[0][t]);
      acc[1][t] = wmma_bf16(af[1], bfr, acc[1][t]);
    }
  }
#pragma unroll
  for (int t = 0; t < 4; ++t) {
    const int ch = n0 + t * 16 + ln;
    const float bvv = bias[ch];
    const int h = ch >> 6, dh = ch & 63;
#pragma unroll
    for (int s = 0; s < 2; ++s) {
      v8bf o;
#pragma unroll
      for (int r = 0; r < 8; ++r) o[r] = (__bf16)(acc[s][t][r] + bvv);
      *(uint4*)(out + (((size_t)b * NH + h) * DH + dh) * NL + m0 + s * 16 +
                off) = *(uint4*)&o;
    }
  }
}

// ---------------------------------------------------------------------------
// Flash attention. One wave = one (b,h) and a 32-query tile (2 groups of 16);
// streams keys in 32-chunks.  Scores computed transposed (S^T = K @ Q^T) so
// each softmax row lives in a lane pair (n, n^16): one shfl_xor(16) per
// reduction.  Every K fragment and V fragment is reused by both query groups.
// O^T = V^T @ P^T accumulated in 2x4 WMMA C tiles (dh x queries).
// Output written as (B, L, 512) bf16 for the final projection.
// ---------------------------------------------------------------------------
__global__ void __launch_bounds__(128) k_attn(
    const __bf16* __restrict__ Qh, const __bf16* __restrict__ Kh,
    const __bf16* __restrict__ Vh, const float* __restrict__ mask,
    __bf16* __restrict__ Ob) {
  const int lane = threadIdx.x & 31;
  const int gw = blockIdx.x * 4 + (threadIdx.x >> 5);
  const int qt = gw & 63;
  const int h  = (gw >> 6) & 7;
  const int b  = gw >> 9;
  const size_t bh = (size_t)b * NH + h;
  const bool hi = lane >= 16;
  const int off = hi ? 8 : 0;
  const int klo = hi ? 16 : 0;
  const int nn = lane & 15;
  const int q0 = qt * 32;

  // Q as B operands (N=queries x K=dh), resident for the whole key loop.
  v16bf qf[2][2];
#pragma unroll
  for (int g = 0; g < 2; ++g) {
    const __bf16* qp = Qh + (bh * NL + q0 + g * 16 + nn) * DH;
    qf[g][0] = ld_frag(qp + klo, qp + klo + 8);
    qf[g][1] = ld_frag(qp + 32 + klo, qp + 32 + klo + 8);
  }

  v8f oacc[2][4] = {};
  float m_run[2] = {-3.0e38f, -3.0e38f};
  float l_run[2] = {0.0f, 0.0f};
  const float* mrow = mask + (size_t)b * NL;

  for (int kb = 0; kb < NL; kb += 32) {
    // K tiles as A operand (M=keys x K=dh): contiguous b128 loads.
    const __bf16* kp0 = Kh + (bh * NL + kb + nn) * DH;
    const __bf16* kp1 = kp0 + 16 * DH;
    v16bf ka00 = ld_frag(kp0 + klo,      kp0 + klo + 8);
    v16bf ka01 = ld_frag(kp0 + 32 + klo, kp0 + 40 + klo);
    v16bf ka10 = ld_frag(kp1 + klo,      kp1 + klo + 8);
    v16bf ka11 = ld_frag(kp1 + 32 + klo, kp1 + 40 + klo);

    // Pull next key chunk toward the WGP while this chunk's WMMAs run.
    if (kb + 32 < NL) {
      __builtin_prefetch(kp0 + 32 * DH, 0, 3);
      __builtin_prefetch(kp1 + 32 * DH, 0, 3);
    }

    // Padding mask for this lane's 16 keys (shared by both query groups).
    const float* mp = mrow + kb;
    float4 ma = *(const float4*)(mp + off);
    float4 mb = *(const float4*)(mp + off + 4);
    float4 mc = *(const float4*)(mp + 16 + off);
    float4 md = *(const float4*)(mp + 20 + off);
    float mv0[8] = {ma.x, ma.y, ma.z, ma.w, mb.x, mb.y, mb.z, mb.w};
    float mv1[8] = {mc.x, mc.y, mc.z, mc.w, md.x, md.y, md.z, md.w};
    float madd0[8], madd1[8];
#pragma unroll
    for (int r = 0; r < 8; ++r) {
      madd0[r] = (mv0[r] <= 0.5f) ? -1.0e9f : 0.0f;
      madd1[r] = (mv1[r] <= 0.5f) ? -1.0e9f : 0.0f;
    }

    v16bf pb[2];
#pragma unroll
    for (int g = 0; g < 2; ++g) {
      v8f c0 = {}, c1 = {};
      c0 = wmma_bf16(ka00, qf[g][0], c0);
      c0 = wmma_bf16(ka01, qf[g][1], c0);
      c1 = wmma_bf16(ka10, qf[g][0], c1);
      c1 = wmma_bf16(ka11, qf[g][1], c1);

      float s0[8], s1[8];
      float mx = -3.0e38f;
#pragma unroll
      for (int r = 0; r < 8; ++r) {
        s0[r] = c0[r] * SCALE + madd0[r];
        s1[r] = c1[r] * SCALE + madd1[r];
        mx = fmaxf(mx, fmaxf(s0[r], s1[r]));
      }
      mx = fmaxf(mx, __shfl_xor(mx, 16, 32));
      const float newm = fmaxf(m_run[g], mx);
      const float corr = __builtin_exp2f((m_run[g] - newm) * L2E);

      float p0[8], p1[8];
      float lsum = 0.0f;
#pragma unroll
      for (int r = 0; r < 8; ++r) {
        p0[r] = __builtin_exp2f((s0[r] - newm) * L2E);
        p1[r] = __builtin_exp2f((s1[r] - newm) * L2E);
        lsum += p0[r] + p1[r];
      }
      lsum += __shfl_xor(lsum, 16, 32);
      l_run[g] = l_run[g] * corr + lsum;
      m_run[g] = newm;

#pragma unroll
      for (int t = 0; t < 4; ++t)
#pragma unroll
        for (int r = 0; r < 8; ++r) oacc[g][t][r] *= corr;

      // Build P^T as a B operand (K=32 keys x N=16 queries): swap half-tiles
      // between the lane pair, then pack to bf16 in K order.
      float pl[8], ph[8];
#pragma unroll
      for (int r = 0; r < 8; ++r) {
        const float send = hi ? p0[r] : p1[r];
        const float recv = __shfl_xor(send, 16, 32);
        pl[r] = hi ? recv : p0[r];
        ph[r] = hi ? p1[r] : recv;
      }
#pragma unroll
      for (int r = 0; r < 8; ++r) {
        pb[g][r]     = (__bf16)pl[r];
        pb[g][8 + r] = (__bf16)ph[r];
      }
    }

    // O^T += V^T(16dh x 32keys) @ P^T(32keys x 16q); V frag shared by groups.
#pragma unroll
    for (int t = 0; t < 4; ++t) {
      const __bf16* vp = Vh + (bh * DH + t * 16 + nn) * NL + kb;
      v16bf va = ld_frag(vp + klo, vp + klo + 8);
      oacc[0][t] = wmma_bf16(va, pb[0], oacc[0][t]);
      oacc[1][t] = wmma_bf16(va, pb[1], oacc[1][t]);
    }
  }

#pragma unroll
  for (int g = 0; g < 2; ++g) {
    const float inv = 1.0f / l_run[g];
#pragma unroll
    for (int t = 0; t < 4; ++t) {
      v8bf o;
#pragma unroll
      for (int r = 0; r < 8; ++r) o[r] = (__bf16)(oacc[g][t][r] * inv);
      *(uint4*)(Ob + ((size_t)b * NL + q0 + g * 16 + nn) * ND + h * DH +
                t * 16 + off) = *(uint4*)&o;
    }
  }
}

// ---------------------------------------------------------------------------
// Output projection (transposed orientation): out(B,512,L) f32.
// A = Ob (M=pos x K=ch), B = Wout (N=outch x K=ch). Per wave: 32x64 tile.
// ---------------------------------------------------------------------------
__global__ void __launch_bounds__(128) k_proj_out(
    const __bf16* __restrict__ Wb, const __bf16* __restrict__ Ob,
    const float* __restrict__ bias, float* __restrict__ out) {
  const int lane = threadIdx.x & 31;
  const int gw = blockIdx.x * 4 + (threadIdx.x >> 5);
  const int b = gw >> 9;
  const int rem = gw & 511;
  const int m0 = (rem >> 3) * 32;  // 64 position tiles of 32
  const int n0 = (rem & 7) * 64;   // 8 channel tiles of 64
  const bool hi = lane >= 16;
  const int off = hi ? 8 : 0;
  const int klo = hi ? 16 : 0;
  const int ln = lane & 15;

  v8f acc[2][4] = {};
  for (int kk = 0; kk < ND; kk += 32) {
    v16bf af[2];
#pragma unroll
    for (int s = 0; s < 2; ++s) {
      const __bf16* ap = Ob + ((size_t)b * NL + m0 + s * 16 + ln) * ND + kk;
      af[s] = ld_frag(ap + klo, ap + klo + 8);
    }
#pragma unroll
    for (int t = 0; t < 4; ++t) {
      const __bf16* bp = Wb + (size_t)(n0 + t * 16 + ln) * ND + kk + klo;
      v16bf bfr = ld_frag(bp, bp + 8);
      acc[0][t] = wmma_bf16(af[0], bfr, acc[0][t]);
      acc[1][t] = wmma_bf16(af[1], bfr, acc[1][t]);
    }
  }
#pragma unroll
  for (int t = 0; t < 4; ++t) {
    const int ch = n0 + t * 16 + ln;
    const float bvv = bias[ch];
#pragma unroll
    for (int s = 0; s < 2; ++s) {
      float* op = out + ((size_t)b * ND + ch) * NL + m0 + s * 16 + off;
      float4 lo = {acc[s][t][0] + bvv, acc[s][t][1] + bvv, acc[s][t][2] + bvv,
                   acc[s][t][3] + bvv};
      float4 hi4 = {acc[s][t][4] + bvv, acc[s][t][5] + bvv, acc[s][t][6] + bvv,
                    acc[s][t][7] + bvv};
      *(float4*)op = lo;
      *(float4*)(op + 4) = hi4;
    }
  }
}

// ---------------------------------------------------------------------------
extern "C" void kernel_launch(void* const* d_in, const int* in_sizes, int n_in,
                              void* d_out, int out_size, void* d_ws,
                              size_t ws_size, hipStream_t stream) {
  const float* q    = (const float*)d_in[0];
  const float* k    = (const float*)d_in[1];
  const float* v    = (const float*)d_in[2];
  const float* mask = (const float*)d_in[3];
  const float* Wq   = (const float*)d_in[4];
  const float* bq   = (const float*)d_in[5];
  const float* Wk   = (const float*)d_in[6];
  const float* bk   = (const float*)d_in[7];
  const float* Wv   = (const float*)d_in[8];
  const float* bv   = (const float*)d_in[9];
  const float* Wo   = (const float*)d_in[10];
  const float* bo   = (const float*)d_in[11];

  char* ws = (char*)d_ws;
  const size_t XT_B = (size_t)NB * NL * ND * 2;  // 8 MiB per bf16 tensor
  const size_t W_B  = (size_t)ND * ND * 2;       // 512 KiB per bf16 weight
  __bf16* xtq = (__bf16*)(ws);
  __bf16* xtk = (__bf16*)(ws + XT_B);
  __bf16* xtv = (__bf16*)(ws + 2 * XT_B);
  __bf16* wqb = (__bf16*)(ws + 3 * XT_B);
  __bf16* wkb = (__bf16*)(ws + 3 * XT_B + W_B);
  __bf16* wvb = (__bf16*)(ws + 3 * XT_B + 2 * W_B);
  __bf16* wob = (__bf16*)(ws + 3 * XT_B + 3 * W_B);
  __bf16* qh  = (__bf16*)(ws + 3 * XT_B + 4 * W_B);
  __bf16* kh  = (__bf16*)(ws + 4 * XT_B + 4 * W_B);
  __bf16* vh  = (__bf16*)(ws + 5 * XT_B + 4 * W_B);
  __bf16* obf = (__bf16*)(ws + 6 * XT_B + 4 * W_B);

  k_cvt<<<512, 256, 0, stream>>>(Wq, wqb, ND * ND);
  k_cvt<<<512, 256, 0, stream>>>(Wk, wkb, ND * ND);
  k_cvt<<<512, 256, 0, stream>>>(Wv, wvb, ND * ND);
  k_cvt<<<512, 256, 0, stream>>>(Wo, wob, ND * ND);

  dim3 tb(32, 8, 1), tg(NL / 32, ND / 32, NB);
  k_transpose<<<tg, tb, 0, stream>>>(q, xtq);
  k_transpose<<<tg, tb, 0, stream>>>(k, xtk);
  k_transpose<<<tg, tb, 0, stream>>>(v, xtv);

  k_proj_qk<<<512, 128, 0, stream>>>(wqb, xtq, bq, qh);
  k_proj_qk<<<512, 128, 0, stream>>>(wkb, xtk, bk, kh);
  k_proj_v <<<512, 128, 0, stream>>>(wvb, xtv, bv, vh);

  k_attn<<<512, 128, 0, stream>>>(qh, kh, vh, mask, obf);

  k_proj_out<<<512, 128, 0, stream>>>(wob, obf, bo, (float*)d_out);

  (void)in_sizes; (void)n_in; (void)out_size; (void)ws_size;
}
